// LTFE_30846455120453
// MI455X (gfx1250) — matrix operand order repaired and checked by reading, not compiled, not run on hardware
//
#include <hip/hip_runtime.h>
#include <hip/hip_bf16.h>
#include <cmath>

// ---------------- problem constants ----------------
#define BB    8
#define CC    1024
#define HH    32
#define WW    32
#define HWSZ  1024          // 32*32
#define TT    4
#define HID   256
#define G4    1024          // 4*HID
#define CHW   (CC*HWSZ)     // 1,048,576
#define BCHW  (BB*CHW)      // 8,388,608

// ---------------- WMMA types ----------------
typedef __bf16 v16bf __attribute__((ext_vector_type(16)));
typedef float  v8f   __attribute__((ext_vector_type(8)));
typedef unsigned short u16x8 __attribute__((ext_vector_type(8)));

__device__ __forceinline__ v16bf make_frag(u16x8 lo, u16x8 hi) {
    union { u16x8 u[2]; v16bf v; } x;
    x.u[0] = lo; x.u[1] = hi;
    return x.v;
}

__device__ __forceinline__ unsigned short f2bf(float f) {
    unsigned u = __float_as_uint(f);
    unsigned r = u + 0x7FFFu + ((u >> 16) & 1u);   // round-to-nearest-even
    return (unsigned short)(r >> 16);
}

__device__ __forceinline__ float sigm(float x) { return 1.0f / (1.0f + __expf(-x)); }

struct Taps { float k[11]; int ks; float alpha; };

// ---------------- utility ----------------
__global__ void k_zero(float* p, int n) {
    int i = blockIdx.x * 256 + threadIdx.x;
    if (i < n) p[i] = 0.0f;
}

// fusion_W[:, HID:]  ->  bf16 row-major [o][c], ld=1024
__global__ void k_cvt_w2(const float* __restrict__ fw, unsigned short* __restrict__ w2) {
    int i = blockIdx.x * 256 + threadIdx.x;          // 1M threads
    int o = i >> 10, c = i & 1023;
    w2[i] = f2bf(fw[(size_t)o * (CC + HID) + HID + c]);
}

// ---------------- diffusion: separable reflect-pad blur ----------------
// out = blurH(Fin + alpha*Nz)
__global__ void k_blur_h(const float* __restrict__ Fin, const float* __restrict__ Nz,
                         float* __restrict__ outp, Taps tp) {
    size_t idx = (size_t)blockIdx.x * 256 + threadIdx.x;   // < BCHW
    int w = idx & 31;
    int h = (idx >> 5) & 31;
    size_t plane = idx >> 10;
    const float* Fb = Fin + (plane << 10);
    const float* Nb = Nz  + (plane << 10);
    int p = tp.ks >> 1;
    float acc = 0.0f;
    for (int i = 0; i < tp.ks; ++i) {
        int hh = h + i - p;
        hh = hh < 0 ? -hh : (hh > 31 ? 62 - hh : hh);
        acc += tp.k[i] * (Fb[hh * 32 + w] + tp.alpha * Nb[hh * 32 + w]);
    }
    outp[idx] = acc;
}

__global__ void k_blur_w(const float* __restrict__ tin, float* __restrict__ outp, Taps tp) {
    size_t idx = (size_t)blockIdx.x * 256 + threadIdx.x;
    int w = idx & 31;
    size_t rowbase = idx & ~31ull;
    const float* R = tin + rowbase;
    int p = tp.ks >> 1;
    float acc = 0.0f;
    for (int i = 0; i < tp.ks; ++i) {
        int ww = w + i - p;
        ww = ww < 0 ? -ww : (ww > 31 ? 62 - ww : ww);
        acc += tp.k[i] * R[ww];
    }
    outp[idx] = acc;
}

// ---------------- pooled mean over HW ----------------
__global__ void k_pool(const float* __restrict__ Ft, float* __restrict__ pooled) {
    __shared__ float sd[256];
    const float* P = Ft + ((size_t)blockIdx.x << 10);
    int t = threadIdx.x;
    sd[t] = P[t] + P[t + 256] + P[t + 512] + P[t + 768];
    __syncthreads();
    for (int o = 128; o > 0; o >>= 1) {
        if (t < o) sd[t] += sd[t + o];
        __syncthreads();
    }
    if (t == 0) pooled[blockIdx.x] = sd[0] * (1.0f / 1024.0f);
}

// ---------------- LSTM ----------------
__global__ void k_lstm_gates(const float* __restrict__ x, const float* __restrict__ h,
                             const float* __restrict__ Wih, const float* __restrict__ Whh,
                             const float* __restrict__ bih, const float* __restrict__ bhh,
                             float* __restrict__ g) {
    int i = blockIdx.x * 256 + threadIdx.x;          // B*1024
    int b = i >> 10, j = i & 1023;
    float acc = bih[j] + bhh[j];
    const float* xr = x + b * CC;
    const float* wr = Wih + (size_t)j * CC;
    for (int c = 0; c < CC; ++c) acc += xr[c] * wr[c];
    const float* hr = h + b * HID;
    const float* vr = Whh + (size_t)j * HID;
    for (int k = 0; k < HID; ++k) acc += hr[k] * vr[k];
    g[i] = acc;
}

__global__ void k_lstm_update(const float* __restrict__ g, float* __restrict__ h,
                              float* __restrict__ c) {
    int i = blockIdx.x * 256 + threadIdx.x;          // B*256
    int b = i >> 8, j = i & 255;
    const float* gb = g + b * G4;
    float gi = sigm(gb[j]);
    float gf = sigm(gb[HID + j]);
    float gg = tanhf(gb[2 * HID + j]);
    float go = sigm(gb[3 * HID + j]);
    float cn = gf * c[i] + gi * gg;
    c[i] = cn;
    h[i] = go * tanhf(cn);
}

// s[b][o] = h_t . fusion_W[o,:HID] + fusion_b[o]
__global__ void k_sbias(const float* __restrict__ h, const float* __restrict__ fw,
                        const float* __restrict__ fb, float* __restrict__ s) {
    int i = blockIdx.x * 256 + threadIdx.x;          // B*1024
    int b = i >> 10, o = i & 1023;
    float acc = fb[o];
    const float* hr = h + b * HID;
    const float* wr = fw + (size_t)o * (CC + HID);
    for (int k = 0; k < HID; ++k) acc += hr[k] * wr[k];
    s[i] = acc;
}

// ---------------- transpose F_t[b][c][hw] -> bf16 Ftb[b][hw][c] ----------------
__global__ void k_transpose_cvt(const float* __restrict__ Ft, unsigned short* __restrict__ Ftb) {
    __shared__ float tile[32][33];
    int b  = blockIdx.z;
    int c0 = blockIdx.y * 32;
    int n0 = blockIdx.x * 32;
    int tx = threadIdx.x & 31, ty = threadIdx.x >> 5;   // ty in 0..7
    const float* src = Ft + ((size_t)b << 20);
    unsigned short* dst = Ftb + ((size_t)b << 20);
#pragma unroll
    for (int it = 0; it < 4; ++it) {
        int cl = ty + it * 8;
        tile[cl][tx] = src[(size_t)(c0 + cl) * 1024 + n0 + tx];
    }
    __syncthreads();
#pragma unroll
    for (int it = 0; it < 4; ++it) {
        int nl = ty + it * 8;
        dst[(size_t)(n0 + nl) * 1024 + c0 + tx] = f2bf(tile[tx][nl]);
    }
}

// ---------------- fusion GEMM + relu-mean epilogue (WMMA bf16) ----------------
// Hp[b][o] += (1/1024) * sum_n relu( (W2 @ Ftb^T)[o,n] + s[b,o] )
// grid: (8 nblocks, 16 mblocks, 8 batch); block = 128 (4 waves); wave tile 64x32
__global__ void __launch_bounds__(128)
k_fusion_wmma(const unsigned short* __restrict__ Abf,   // W2 bf16 [1024][1024]
              const unsigned short* __restrict__ Bbf,   // Ftb bf16 [B][1024 n][1024 c]
              const float* __restrict__ sbias,          // [B][1024]
              float* __restrict__ Hp) {                 // [B][1024]
    const int b     = blockIdx.z;
    const int wave  = threadIdx.x >> 5;
    const int lane  = threadIdx.x & 31;
    const int l15   = lane & 15;
    const int half  = lane >> 4;
    const int nBase = blockIdx.x * 128 + wave * 32;
    const int oBase = blockIdx.y * 64;

    const unsigned short* Bb = Bbf + ((size_t)b << 20);

    v8f acc[4][2] = {};

    for (int k0 = 0; k0 < 1024; k0 += 32) {
        v16bf aF[4], bF[2];
        // A (16x32, 16-bit): lane l15 = row M; K = (e<8?0:16) + half*8 + (e&7)
#pragma unroll
        for (int mt = 0; mt < 4; ++mt) {
            const unsigned short* p =
                Abf + (size_t)(oBase + mt * 16 + l15) * 1024 + k0 + half * 8;
            aF[mt] = make_frag(*(const u16x8*)p, *(const u16x8*)(p + 16));
        }
        // B (32x16, 16-bit): lane l15 = col N; K = (e<8?0:16) + half*8 + (e&7)
#pragma unroll
        for (int nt = 0; nt < 2; ++nt) {
            const unsigned short* p =
                Bb + (size_t)(nBase + nt * 16 + l15) * 1024 + k0 + half * 8;
            bF[nt] = make_frag(*(const u16x8*)p, *(const u16x8*)(p + 16));
        }
#pragma unroll
        for (int mt = 0; mt < 4; ++mt)
#pragma unroll
            for (int nt = 0; nt < 2; ++nt)
                acc[mt][nt] = __builtin_amdgcn_wmma_f32_16x16x32_bf16(
                    false, aF[mt], false, bF[nt], (short)0, acc[mt][nt], false, false);
    }

    // epilogue: +bias, relu, reduce over the 32 n of this wave, atomic into Hp
    const float inv = 1.0f / 1024.0f;
#pragma unroll
    for (int mt = 0; mt < 4; ++mt) {
#pragma unroll
        for (int r = 0; r < 8; ++r) {
            int o = oBase + mt * 16 + r + half * 8;   // D layout: VGPR r -> M=r(+8)
            float sb = sbias[b * 1024 + o];
            float v0 = acc[mt][0][r] + sb; v0 = v0 > 0.0f ? v0 : 0.0f;
            float v1 = acc[mt][1][r] + sb; v1 = v1 > 0.0f ? v1 : 0.0f;
            float s = v0 + v1;
            s += __shfl_xor(s, 1, 32);
            s += __shfl_xor(s, 2, 32);
            s += __shfl_xor(s, 4, 32);
            s += __shfl_xor(s, 8, 32);
            if (l15 == 0) atomicAdd(&Hp[b * 1024 + o], s * inv);
        }
    }
}

// ---------------- tau ----------------
__global__ void k_tau_part(const float* __restrict__ Hp, float* __restrict__ traw) {
    __shared__ float sd[256];
    const float* hp = Hp + blockIdx.x * 1024;
    int t = threadIdx.x;
    float a = hp[t], b2 = hp[t + 256], c2 = hp[t + 512], d2 = hp[t + 768];
    sd[t] = a * a + b2 * b2 + c2 * c2 + d2 * d2;
    __syncthreads();
    for (int o = 128; o > 0; o >>= 1) {
        if (t < o) sd[t] += sd[t + o];
        __syncthreads();
    }
    if (t == 0) traw[blockIdx.x] = sqrtf(sd[0]);
}

__global__ void k_tau_fin(const float* __restrict__ traw, float* __restrict__ tn) {
    __shared__ float m;
    if (threadIdx.x == 0) {
        float mx = traw[0];
        for (int i = 1; i < BB; ++i) mx = fmaxf(mx, traw[i]);
        m = mx;
    }
    __syncthreads();
    if (threadIdx.x < BB) tn[threadIdx.x] = traw[threadIdx.x] / (m + 1e-8f);
}

// ---------------- ODE MLP ----------------
__global__ void k_ode1(const float* __restrict__ h, const float* __restrict__ W1,
                       const float* __restrict__ b1, float* __restrict__ z) {
    int i = blockIdx.x * 256 + threadIdx.x;          // B*128
    int b = i >> 7, j = i & 127;
    float acc = b1[j];
    const float* hr = h + b * HID;
    const float* wr = W1 + (size_t)j * HID;
    for (int k = 0; k < HID; ++k) acc += hr[k] * wr[k];
    z[i] = acc > 0.0f ? acc : 0.0f;
}

__global__ void k_ode2(const float* __restrict__ z, const float* __restrict__ W2,
                       const float* __restrict__ b2, float* __restrict__ dW) {
    int i = blockIdx.x * 256 + threadIdx.x;          // B*9216
    int b = i / 9216, m = i % 9216;
    float acc = b2[m];
    const float* zr = z + b * 128;
    const float* wr = W2 + (size_t)m * 128;
    for (int j = 0; j < 128; ++j) acc += zr[j] * wr[j];
    dW[i] = acc;
}

// ---------------- per-sample depthwise 3x3 + residual ----------------
__global__ void k_fhat(const float* __restrict__ F0, const float* __restrict__ W0,
                       const float* __restrict__ dW, const float* __restrict__ tn,
                       float* __restrict__ out) {
    int blk = blockIdx.x;                 // b*1024 + c
    int b = blk >> 10, c = blk & 1023;
    float tb = tn[b];
    float w[9];
#pragma unroll
    for (int p = 0; p < 9; ++p)
        w[p] = W0[c * 9 + p] + tb * dW[b * 9216 + c * 9 + p];
    const float* Fp = F0 + ((size_t)blk << 10);
    float* op = out + ((size_t)blk << 10);
    for (int px = threadIdx.x; px < 1024; px += 256) {
        int h = px >> 5, wc = px & 31;
        float acc = Fp[px];               // + F_0 residual
#pragma unroll
        for (int di = 0; di < 3; ++di) {
            int hh = h + di - 1;
            if (hh < 0 || hh > 31) continue;
#pragma unroll
            for (int dj = 0; dj < 3; ++dj) {
                int ww = wc + dj - 1;
                if (ww < 0 || ww > 31) continue;
                acc += w[di * 3 + dj] * Fp[hh * 32 + ww];
            }
        }
        op[px] = acc;
    }
}

// ---------------- launcher ----------------
extern "C" void kernel_launch(void* const* d_in, const int* in_sizes, int n_in,
                              void* d_out, int out_size, void* d_ws, size_t ws_size,
                              hipStream_t stream) {
    const float* F0   = (const float*)d_in[0];
    const float* nz   = (const float*)d_in[1];
    const float* Wih  = (const float*)d_in[2];
    const float* Whh  = (const float*)d_in[3];
    const float* bih  = (const float*)d_in[4];
    const float* bhh  = (const float*)d_in[5];
    const float* fw   = (const float*)d_in[6];
    const float* fb   = (const float*)d_in[7];
    const float* oW1  = (const float*)d_in[8];
    const float* ob1  = (const float*)d_in[9];
    const float* oW2  = (const float*)d_in[10];
    const float* ob2  = (const float*)d_in[11];
    const float* W0   = (const float*)d_in[12];
    float* out = (float*)d_out;

    // workspace carve-up (all 256B aligned)
    char* ws = (char*)d_ws;
    size_t off = 0;
    auto take = [&](size_t bytes) {
        char* p = ws + off;
        off = (off + bytes + 255) & ~(size_t)255;
        return p;
    };
    float*          tmpH   = (float*)take((size_t)BCHW * 4);        // 33.5 MB
    unsigned short* Ftb    = (unsigned short*)take((size_t)BCHW * 2); // 16.8 MB
    unsigned short* W2bf   = (unsigned short*)take((size_t)CC * CC * 2); // 2 MB
    float*          pooled = (float*)take((size_t)BB * CC * 4);
    float*          gates  = (float*)take((size_t)BB * G4 * 4);
    float*          hcbuf  = (float*)take((size_t)2 * BB * HID * 4);
    float*          sbias  = (float*)take((size_t)BB * CC * 4);
    float*          Hp     = (float*)take((size_t)BB * CC * 4);
    float*          traw   = (float*)take(64);
    float*          tnrm   = (float*)take(64);
    float*          zbuf   = (float*)take((size_t)BB * 128 * 4);
    float*          dWb    = (float*)take((size_t)BB * CC * 9 * 4);
    float* hbuf = hcbuf;
    float* cbuf = hcbuf + BB * HID;
    (void)ws_size; (void)in_sizes; (void)n_in; (void)out_size;

    // one-time per call: zero LSTM state, convert W2 to bf16
    k_zero<<<(2 * BB * HID + 255) / 256, 256, 0, stream>>>(hcbuf, 2 * BB * HID);
    k_cvt_w2<<<CC * CC / 256, 256, 0, stream>>>(fw, W2bf);

    const size_t BLKS = BCHW / 256;   // 32768

    for (int tt = 0; tt < TT; ++tt) {
        const int t = tt + 1;
        // host-computed Gaussian taps (deterministic constants)
        Taps tp{};
        double sigma = std::pow(1.2, (double)t);
        int ks = 2 * (int)std::ceil(2.0 * sigma) + 1;
        if (ks < 3) ks = 3;
        if ((ks & 1) == 0) ks += 1;
        double half = (ks - 1) * 0.5, sum = 0.0;
        for (int i = 0; i < ks; ++i) {
            double x = (i - half) / sigma;
            tp.k[i] = (float)std::exp(-0.5 * x * x);
            sum += tp.k[i];
        }
        for (int i = 0; i < ks; ++i) tp.k[i] = (float)(tp.k[i] / sum);
        tp.ks = ks;
        tp.alpha = (float)(0.2 * std::exp(-0.2 * (double)t));

        const float* Fprev = (tt == 0) ? F0 : out + (size_t)(tt - 1) * BCHW;
        float* Ft = out + (size_t)tt * BCHW;                     // F_seq slot = storage

        // diffusion: F_t = blur(F_{t-1} + alpha*noise_t)  (separable, reflect)
        k_blur_h<<<BLKS, 256, 0, stream>>>(Fprev, nz + (size_t)tt * BCHW, tmpH, tp);
        k_blur_w<<<BLKS, 256, 0, stream>>>(tmpH, Ft, tp);

        // LSTM step
        k_pool<<<BB * CC, 256, 0, stream>>>(Ft, pooled);
        k_lstm_gates<<<BB * G4 / 256, 256, 0, stream>>>(pooled, hbuf, Wih, Whh, bih, bhh, gates);
        k_lstm_update<<<BB * HID / 256, 256, 0, stream>>>(gates, hbuf, cbuf);

        // fusion GEMM -> Hp (fused relu-mean), via bf16 WMMA out of L2
        k_sbias<<<BB * CC / 256, 256, 0, stream>>>(hbuf, fw, fb, sbias);
        k_transpose_cvt<<<dim3(32, 32, BB), 256, 0, stream>>>(Ft, Ftb);
        k_zero<<<BB * CC / 256, 256, 0, stream>>>(Hp, BB * CC);
        k_fusion_wmma<<<dim3(8, 16, BB), 128, 0, stream>>>(W2bf, Ftb, sbias, Hp);

        // tau
        k_tau_part<<<BB, 256, 0, stream>>>(Hp, traw);
        k_tau_fin<<<1, 32, 0, stream>>>(traw, tnrm);

        // ODE MLP -> dW
        k_ode1<<<BB * 128 / 256, 256, 0, stream>>>(hbuf, oW1, ob1, zbuf);
        k_ode2<<<BB * 9216 / 256, 256, 0, stream>>>(zbuf, oW2, ob2, dWb);

        // per-sample depthwise 3x3 + residual -> F_hat slot
        k_fhat<<<BB * CC, 256, 0, stream>>>(F0, W0, dWb, tnrm,
                                            out + (size_t)TT * BCHW + (size_t)tt * BCHW);
    }
}
